// MEModule_89215060673111
// MI455X (gfx1250) — compile-verified, matrix operand order checked by reading, not run on hardware
//
#include <hip/hip_runtime.h>
#include <hip/hip_bf16.h>
#include <cstdint>

namespace {
constexpr int Cc = 256;
constexpr int CR = 16;
constexpr int Rr = 16;
constexpr int Tt = 8;
constexpr int NBb = 8;
constexpr int NTt = 64;          // NB * T
constexpr int Hh = 56, Ww = 56, HW = 56 * 56;
constexpr float EPSf = 1e-5f;
}

typedef __attribute__((ext_vector_type(16))) _Float16 v16h;
typedef __attribute__((ext_vector_type(8)))  float    v8f;

// ---------------------------------------------------------------------------
// K1: per-(nt,c) spatial mean of x -> pool[NT*C]   (float4 streaming reduce)
// ---------------------------------------------------------------------------
__global__ void k_pool(const float* __restrict__ x, float* __restrict__ pool) {
  const int b = blockIdx.x;                       // nt*C + c
  const float4* p = (const float4*)(x + (size_t)b * HW);
  float s = 0.f;
  for (int i = threadIdx.x; i < HW / 4; i += 256) {
    float4 v = p[i];
    s += (v.x + v.y) + (v.z + v.w);
  }
#pragma unroll
  for (int off = 16; off > 0; off >>= 1) s += __shfl_down(s, off, 32);
  __shared__ float red[8];
  if ((threadIdx.x & 31) == 0) red[threadIdx.x >> 5] = s;
  __syncthreads();
  if (threadIdx.x == 0) {
    float t = 0.f;
#pragma unroll
    for (int i = 0; i < 8; ++i) t += red[i];
    pool[b] = t * (1.0f / HW);
  }
}

// ---------------------------------------------------------------------------
// K2: channel-excitation branch (tiny) -> sg1[nt][c] = 1 + (sigmoid(bn)-0.5)
//     also folds BN1 / BN3 into scale/bias constants.
// ---------------------------------------------------------------------------
__global__ void k_gate1(const float* __restrict__ pool,
                        const float* __restrict__ w_sq,
                        const float* __restrict__ w_ac1,
                        const float* __restrict__ w_ex,
                        const float* __restrict__ abn_g, const float* __restrict__ abn_b,
                        const float* __restrict__ abn_m, const float* __restrict__ abn_v,
                        const float* __restrict__ bn1_g, const float* __restrict__ bn1_b,
                        const float* __restrict__ bn1_m, const float* __restrict__ bn1_v,
                        const float* __restrict__ bn3_g, const float* __restrict__ bn3_b,
                        const float* __restrict__ bn3_m, const float* __restrict__ bn3_v,
                        float* __restrict__ sg1,
                        float* __restrict__ scale1, float* __restrict__ bias1,
                        float* __restrict__ scale3, float* __restrict__ bias3) {
  __shared__ float g1[NTt * CR];     // [nt][cr]  squeeze result
  __shared__ float g2r[NTt * CR];    // [nt][cr]  after temporal conv
  const int tid = threadIdx.x;
  for (int o = tid; o < NTt * CR; o += 256) {       // g1 = pool @ w_sq.T
    int nt = o / CR, cr = o % CR;
    float acc = 0.f;
    for (int c = 0; c < Cc; ++c) acc += pool[nt * Cc + c] * w_sq[cr * Cc + c];
    g1[o] = acc;
  }
  __syncthreads();
  for (int o = tid; o < NTt * CR; o += 256) {       // temporal k=3 pad=1 conv over T
    int nt = o / CR, oc = o % CR;
    int n = nt / Tt, t = nt % Tt;
    float acc = 0.f;
    for (int k = 0; k < 3; ++k) {
      int tt = t + k - 1;
      if (tt < 0 || tt >= Tt) continue;
      const float* gin = &g1[(n * Tt + tt) * CR];
      for (int i = 0; i < CR; ++i) acc += gin[i] * w_ac1[(oc * CR + i) * 3 + k];
    }
    g2r[o] = acc;
  }
  __syncthreads();
  for (int o = tid; o < NTt * Cc; o += 256) {       // expand + BN + sigmoid - 0.5
    int nt = o / Cc, c = o % Cc;
    float acc = 0.f;
    const float* gi = &g2r[nt * CR];
    for (int i = 0; i < CR; ++i) acc += gi[i] * w_ex[c * CR + i];
    float sc = abn_g[c] * rsqrtf(abn_v[c] + EPSf);
    float v = (acc - abn_m[c]) * sc + abn_b[c];
    float sig = 1.0f / (1.0f + __expf(-v));
    sg1[o] = 0.5f + sig;                            // == 1 + (sigmoid - 0.5)
  }
  if (tid < Rr) {
    float s = bn1_g[tid] * rsqrtf(bn1_v[tid] + EPSf);
    scale1[tid] = s;
    bias1[tid] = bn1_b[tid] - bn1_m[tid] * s;
  }
  for (int c = tid; c < Cc; c += 256) {
    float s = bn3_g[c] * rsqrtf(bn3_v[c] + EPSf);
    scale3[c] = s;
    bias3[c] = bn3_b[c] - bn3_m[c] * s;
  }
}

// ---------------------------------------------------------------------------
// K3: WMMA channel-reduce GEMM.
//   bottleneck[nt, n, p] = BN1( sum_k  x[nt,k,p]*sg1[nt,k] * w_c1[n,k] )
//   One wave per 16-pixel tile; K=256 fully unrolled into 8 back-to-back
//   v_wmma_f32_16x16x32_f16 so loads pipeline across matrix issues.
// ---------------------------------------------------------------------------
__global__ void __launch_bounds__(128)
k_bottleneck(const float* __restrict__ x, const float* __restrict__ sg1,
             const float* __restrict__ w_c1,
             const float* __restrict__ scale1, const float* __restrict__ bias1,
             float* __restrict__ bt) {
  const int wave = threadIdx.x >> 5;               // 4 waves / block
  const int lane = threadIdx.x & 31;
  const int nt   = blockIdx.x / 49;                // 49 blocks * 4 waves = 196 tiles
  const int tile = (blockIdx.x % 49) * 4 + wave;
  const int p0   = tile * 16;

  const int m     = lane & 15;                     // pixel row of A / out-chan of B
  const int khalf = (lane >> 4) << 3;              // K sub-block 0 or 8
  const float* xp   = x + ((size_t)nt * Cc) * HW + (size_t)(p0 + m);
  const float* sg   = sg1 + nt * Cc;
  const float* wrow = w_c1 + m * Cc;               // w_c1[n = lane&15][*]

  v8f acc = {};
#pragma unroll
  for (int kb = 0; kb < Cc; kb += 32) {
    v16h a, b;
#pragma unroll
    for (int i = 0; i < 16; ++i) {
      const int k = kb + ((i >> 3) << 4) + khalf + (i & 7);
      a[i] = (_Float16)(xp[(size_t)k * HW] * sg[k]);   // gated x_p2, A[m][k]
      b[i] = (_Float16)(wrow[k]);                      // B[k][n] = w_c1[n][k]
    }
    acc = __builtin_amdgcn_wmma_f32_16x16x32_f16(false, a, false, b,
                                                 (short)0, acc, false, false);
  }
  // D layout: lane = n + 16*(m>=8); acc[r] = D[mb+r][n]
  const int n  = lane & 15;
  const int mb = (lane >> 4) << 3;
  const float s1 = scale1[n], b1 = bias1[n];
  float* outp = bt + ((size_t)nt * Rr + n) * HW + p0 + mb;
#pragma unroll
  for (int r2 = 0; r2 < 8; ++r2) outp[r2] = acc[r2] * s1 + b1;
}

// ---------------------------------------------------------------------------
// K4: dfp[f,r,p] = (t<7) ? depthwise3x3(bt[f+1]) - bt[f] : 0
// ---------------------------------------------------------------------------
__global__ void k_dfp(const float* __restrict__ bt, const float* __restrict__ w_c2,
                      float* __restrict__ dfp) {
  size_t idx = (size_t)blockIdx.x * blockDim.x + threadIdx.x;
  if (idx >= (size_t)NTt * Rr * HW) return;
  int p = (int)(idx % HW);
  int r = (int)((idx / HW) % Rr);
  int f = (int)(idx / ((size_t)HW * Rr));
  int t = f & 7;
  float v = 0.f;
  if (t < 7) {
    int y = p / Ww, xq = p - y * Ww;
    const float* src = bt + ((size_t)(f + 1) * Rr + r) * HW;
    float acc = 0.f;
#pragma unroll
    for (int dy = -1; dy <= 1; ++dy)
#pragma unroll
      for (int dx = -1; dx <= 1; ++dx) {
        int yy = y + dy, xx = xq + dx;
        if (yy >= 0 && yy < Hh && xx >= 0 && xx < Ww)
          acc += src[yy * Ww + xx] * w_c2[(r * 3 + (dy + 1)) * 3 + (dx + 1)];
      }
    v = acc - bt[((size_t)f * Rr + r) * HW + p];
  }
  dfp[idx] = v;
}

// ---------------------------------------------------------------------------
// K5: per (frame,channel) 9 stats: {T, R0, R55, C0, C55, X00, X0w, Xh0, Xhw}
//     mean(conv3x3_pad1(z)) is a linear function of these.
// ---------------------------------------------------------------------------
__global__ void k_stats(const float* __restrict__ dfp, float* __restrict__ stats) {
  const int b = blockIdx.x;                        // f*R + r
  const float* p = dfp + (size_t)b * HW;
  float t = 0, r0 = 0, r1 = 0, c0 = 0, c1 = 0;
  for (int i = threadIdx.x; i < HW; i += 256) {
    float v = p[i];
    t += v;
    int y = i / Ww, xq = i - y * Ww;
    if (y == 0)       r0 += v;
    if (y == Hh - 1)  r1 += v;
    if (xq == 0)      c0 += v;
    if (xq == Ww - 1) c1 += v;
  }
#pragma unroll
  for (int off = 16; off > 0; off >>= 1) {
    t  += __shfl_down(t,  off, 32);
    r0 += __shfl_down(r0, off, 32);
    r1 += __shfl_down(r1, off, 32);
    c0 += __shfl_down(c0, off, 32);
    c1 += __shfl_down(c1, off, 32);
  }
  __shared__ float red[8][5];
  int w = threadIdx.x >> 5;
  if ((threadIdx.x & 31) == 0) {
    red[w][0] = t; red[w][1] = r0; red[w][2] = r1; red[w][3] = c0; red[w][4] = c1;
  }
  __syncthreads();
  if (threadIdx.x == 0) {
    float a0 = 0, a1 = 0, a2 = 0, a3 = 0, a4 = 0;
    for (int i = 0; i < 8; ++i) {
      a0 += red[i][0]; a1 += red[i][1]; a2 += red[i][2]; a3 += red[i][3]; a4 += red[i][4];
    }
    float* s = stats + b * 9;
    s[0] = a0; s[1] = a1; s[2] = a2; s[3] = a3; s[4] = a4;
    s[5] = p[0]; s[6] = p[Ww - 1]; s[7] = p[(Hh - 1) * Ww]; s[8] = p[HW - 1];
  }
}

__device__ __forceinline__ float shiftSum(const float* s, int dy, int dx) {
  float v = s[0];
  if (dy == -1) v -= s[2]; else if (dy == 1) v -= s[1];
  if (dx == -1) v -= s[4]; else if (dx == 1) v -= s[3];
  if (dy != 0 && dx != 0) {
    int yr = (dy == -1) ? 1 : 0;                  // 1 => row 55 excluded
    int xc = (dx == -1) ? 1 : 0;                  // 1 => col 55 excluded
    if (!yr && !xc) v += s[5];
    else if (!yr)   v += s[6];
    else if (!xc)   v += s[7];
    else            v += s[8];
  }
  return v;
}

// ---------------------------------------------------------------------------
// K6: everything downstream of stats (tiny) -> sg2[nt][c] = 1 + (sigmoid-0.5)
// ---------------------------------------------------------------------------
__global__ void k_tail(const float* __restrict__ stats,
                       const float* __restrict__ w_c8s, const float* __restrict__ b_c8s,
                       const float* __restrict__ w_c8t, const float* __restrict__ b_c8t,
                       const float* __restrict__ w_cs,  const float* __restrict__ b_cs,
                       const float* __restrict__ w_cst, const float* __restrict__ b_cst,
                       const float* __restrict__ w_up,  const float* __restrict__ b_up,
                       const float* __restrict__ w_c3,
                       const float* __restrict__ scale3, const float* __restrict__ bias3,
                       float* __restrict__ sg2) {
  __shared__ float d8pre[NTt][CR];
  __shared__ float dspre[NBb][6][CR];
  __shared__ float dd[NBb][CR][4];
  __shared__ float d12[NBb][CR][12];
  __shared__ float e8[NBb][CR][8];
  const int tid = threadIdx.x;

  for (int o = tid; o < NTt * CR; o += 256) {       // mean(conv(dfp,w_c8s)+b)
    int f = o / CR, oc = o % CR;
    float acc = 0.f;
    for (int i = 0; i < CR; ++i) {
      const float* s = stats + (f * CR + i) * 9;
      int wb = (oc * CR + i) * 9;
#pragma unroll
      for (int dy = -1; dy <= 1; ++dy)
#pragma unroll
        for (int dx = -1; dx <= 1; ++dx)
          acc += w_c8s[wb + (dy + 1) * 3 + (dx + 1)] * shiftSum(s, dy, dx);
    }
    d8pre[f][oc] = b_c8s[oc] + acc * (1.0f / HW);
  }
  for (int o = tid; o < NBb * 6 * CR; o += 256) {   // mean(conv(dsec,w_cs)+b)
    int oc = o % CR; int t = (o / CR) % 6; int n = o / (6 * CR);
    int f = n * Tt + t;
    float acc = 0.f;
    for (int i = 0; i < CR; ++i) {
      const float* sa = stats + (f * CR + i) * 9;
      const float* sb = stats + ((f + 1) * CR + i) * 9;
      float s[9];
#pragma unroll
      for (int q = 0; q < 9; ++q) s[q] = sb[q] - sa[q];   // linearity of stats
      int wb = (oc * CR + i) * 9;
#pragma unroll
      for (int dy = -1; dy <= 1; ++dy)
#pragma unroll
        for (int dx = -1; dx <= 1; ++dx)
          acc += w_cs[wb + (dy + 1) * 3 + (dx + 1)] * shiftSum(s, dy, dx);
    }
    dspre[n][t][oc] = b_cs[oc] + acc * (1.0f / HW);
  }
  __syncthreads();
  for (int o = tid; o < NBb * CR * 4; o += 256) {   // temporal convs + d4 + mix
    int tt = o & 3; int oc = (o >> 2) % CR; int n = o / (CR * 4);
    float a8 = b_c8t[oc];                            // stride2 pad1 k3
    for (int k = 0; k < 3; ++k) {
      int tau = 2 * tt - 1 + k;
      if (tau < 0 || tau >= Tt) continue;
      for (int i = 0; i < CR; ++i)
        a8 += w_c8t[(oc * CR + i) * 3 + k] * d8pre[n * Tt + tau][i];
    }
    float as = b_cst[oc];                            // k3 pad0 over 6
    for (int k = 0; k < 3; ++k)
      for (int i = 0; i < CR; ++i)
        as += w_cst[(oc * CR + i) * 3 + k] * dspre[n][tt + k][i];
    int q = 4 * n + tt;                              // torch frame-major cat order
    int n_idx = q % NBb, tb = 2 * (q / NBb);
    float d4v = stats[((n_idx * Tt + tb) * CR + oc) * 9 + 0] * (1.0f / HW);
    dd[n][oc][tt] = (d4v + a8 + as) * (1.0f / 3.0f);
  }
  __syncthreads();
  for (int o = tid; o < NBb * 48 * 4; o += 256) {   // 1x1 up (R->3R) + view(n,R,12)
    int tt = o & 3; int o3 = (o >> 2) % 48; int n = o / (48 * 4);
    float acc = b_up[o3];
    for (int r = 0; r < Rr; ++r) acc += w_up[o3 * Rr + r] * dd[n][r][tt];
    int flat = o3 * 4 + tt;
    d12[n][flat / 12][flat % 12] = acc;
  }
  __syncthreads();
  for (int o = tid; o < NBb * CR; o += 256) {       // select/avg 12 -> 8
    int r = o % CR, n = o / CR;
    const float* v = d12[n][r];
    float* w8 = e8[n][r];
    w8[0] = v[1];
    w8[1] = 0.5f * (v[2] + v[3]);
    w8[2] = v[4];
    w8[3] = 0.5f * (v[5] + v[6]);
    w8[4] = v[7];
    w8[5] = 0.5f * (v[8] + v[9]);
    w8[6] = v[10];
    w8[7] = v[11];
  }
  __syncthreads();
  for (int o = tid; o < NTt * Cc; o += 256) {       // expand R->C + BN3 + sigmoid
    int c = o % Cc, nt = o / Cc;
    int n = nt / Tt, t = nt % Tt;
    float acc = 0.f;
    for (int r = 0; r < Rr; ++r) acc += e8[n][r][t] * w_c3[c * Rr + r];
    float v = acc * scale3[c] + bias3[c];
    float sig = 1.0f / (1.0f + __expf(-v));
    sg2[o] = 0.5f + sig;
  }
}

// ---------------------------------------------------------------------------
// K7: out = x_p2 + x_p2*y = x * sg1 * sg2   (pure streaming, float4)
// ---------------------------------------------------------------------------
__global__ void k_out(const float* __restrict__ x, const float* __restrict__ sg1,
                      const float* __restrict__ sg2, float* __restrict__ out) {
  const size_t total = (size_t)NTt * Cc * (HW / 4);   // float4 count
  const size_t stride = (size_t)gridDim.x * blockDim.x;
  const float4* xv = (const float4*)x;
  float4* ov = (float4*)out;
  for (size_t i = (size_t)blockIdx.x * blockDim.x + threadIdx.x; i < total; i += stride) {
    __builtin_prefetch(&xv[i + stride], 0, 0);        // global_prefetch_b8
    size_t plane = i / (HW / 4);                      // nt*C + c
    float s = sg1[plane] * sg2[plane];
    float4 v = xv[i];
    v.x *= s; v.y *= s; v.z *= s; v.w *= s;
    ov[i] = v;
  }
}

// ---------------------------------------------------------------------------
extern "C" void kernel_launch(void* const* d_in, const int* in_sizes, int n_in,
                              void* d_out, int out_size, void* d_ws, size_t ws_size,
                              hipStream_t stream) {
  (void)in_sizes; (void)n_in; (void)out_size; (void)ws_size;
  const float* x     = (const float*)d_in[0];
  const float* w_sq  = (const float*)d_in[1];
  const float* w_ac1 = (const float*)d_in[2];
  const float* w_ex  = (const float*)d_in[3];
  const float* abn_g = (const float*)d_in[4];
  const float* abn_b = (const float*)d_in[5];
  const float* abn_m = (const float*)d_in[6];
  const float* abn_v = (const float*)d_in[7];
  const float* w_c1  = (const float*)d_in[8];
  const float* bn1_g = (const float*)d_in[9];
  const float* bn1_b = (const float*)d_in[10];
  const float* bn1_m = (const float*)d_in[11];
  const float* bn1_v = (const float*)d_in[12];
  const float* w_c2  = (const float*)d_in[13];
  const float* w_c8s = (const float*)d_in[14];
  const float* b_c8s = (const float*)d_in[15];
  const float* w_c8t = (const float*)d_in[16];
  const float* b_c8t = (const float*)d_in[17];
  const float* w_cs  = (const float*)d_in[18];
  const float* b_cs  = (const float*)d_in[19];
  const float* w_cst = (const float*)d_in[20];
  const float* b_cst = (const float*)d_in[21];
  const float* w_up  = (const float*)d_in[22];
  const float* b_up  = (const float*)d_in[23];
  const float* w_c3  = (const float*)d_in[24];
  const float* bn3_g = (const float*)d_in[25];
  const float* bn3_b = (const float*)d_in[26];
  const float* bn3_m = (const float*)d_in[27];
  const float* bn3_v = (const float*)d_in[28];

  float* ws     = (float*)d_ws;
  float* pool   = ws;                               // 16384
  float* sg1    = pool + NTt * Cc;                  // 16384
  float* sg2    = sg1 + NTt * Cc;                   // 16384
  float* scale1 = sg2 + NTt * Cc;                   // 16
  float* bias1  = scale1 + Rr;                      // 16
  float* scale3 = bias1 + Rr;                       // 256
  float* bias3  = scale3 + Cc;                      // 256
  float* stats  = bias3 + Cc;                       // 64*16*9
  float* bt     = stats + NTt * Rr * 9;             // 64*16*3136
  float* dfp    = bt + (size_t)NTt * Rr * HW;       // 64*16*3136

  k_pool<<<NTt * Cc, 256, 0, stream>>>(x, pool);
  k_gate1<<<1, 256, 0, stream>>>(pool, w_sq, w_ac1, w_ex,
                                 abn_g, abn_b, abn_m, abn_v,
                                 bn1_g, bn1_b, bn1_m, bn1_v,
                                 bn3_g, bn3_b, bn3_m, bn3_v,
                                 sg1, scale1, bias1, scale3, bias3);
  k_bottleneck<<<NTt * 49, 128, 0, stream>>>(x, sg1, w_c1, scale1, bias1, bt);
  {
    size_t tot = (size_t)NTt * Rr * HW;
    k_dfp<<<(int)((tot + 255) / 256), 256, 0, stream>>>(bt, w_c2, dfp);
  }
  k_stats<<<NTt * Rr, 256, 0, stream>>>(dfp, stats);
  k_tail<<<1, 256, 0, stream>>>(stats, w_c8s, b_c8s, w_c8t, b_c8t,
                                w_cs, b_cs, w_cst, b_cst, w_up, b_up,
                                w_c3, scale3, bias3, sg2);
  k_out<<<4096, 256, 0, stream>>>(x, sg1, sg2, (float*)d_out);
}
